// SparseAttention_12472585027939
// MI455X (gfx1250) — matrix operand order repaired and checked by reading, not compiled, not run on hardware
//
#include <hip/hip_runtime.h>

#define B 8
#define L 2048
#define H 8
#define Dh 64
#define BH 64
#define U 40
#define UP 48
#define SCALE 0.125f   // 1/sqrt(64)

typedef __attribute__((ext_vector_type(16))) __bf16 v16bf;
typedef __attribute__((ext_vector_type(8)))  __bf16 v8bf;
typedef __attribute__((ext_vector_type(8)))  float  v8f;

__device__ __forceinline__ unsigned short f2bf(float x){
  unsigned u = __builtin_bit_cast(unsigned, x);
  u += 0x7FFFu + ((u >> 16) & 1u);            // round-to-nearest-even
  return (unsigned short)(u >> 16);
}

// A-fragment (16x32 bf16, MxK): lane holds K = {kb..kb+15}, kb = 16*(lane>=16)
__device__ __forceinline__ v16bf load_a_frag(const unsigned short* base, int half){
  const int kb = half * 16;
  v8bf lo = *(const v8bf*)(base + kb);
  v8bf hi = *(const v8bf*)(base + kb + 8);
  return __builtin_shufflevector(lo, hi, 0,1,2,3,4,5,6,7,8,9,10,11,12,13,14,15);
}
// B-fragment (32x16 bf16, KxN): lane holds 16 contiguous elements along K for column n = lane&15
__device__ __forceinline__ v16bf load_b_frag(const unsigned short* p){
  return *(const v16bf*)p;
}

// ---------------- pack f32 [B,L,H,D] -> bf16 [B*H][L][D] ----------------
__global__ void __launch_bounds__(256) pack_bf16(const float* __restrict__ src,
                                                 unsigned short* __restrict__ dst){
  int g = blockIdx.x * 256 + threadIdx.x;          // one thread per 4 elements
  int d4 = g & 15;
  int l  = (g >> 4) & (L - 1);
  int h  = (g >> 15) & (H - 1);
  int b  = g >> 18;
  float4 v = *(const float4*)(src + (((long)(b*L + l)*H + h)*Dh + d4*4));
  ushort4 o; o.x = f2bf(v.x); o.y = f2bf(v.y); o.z = f2bf(v.z); o.w = f2bf(v.w);
  *(ushort4*)(dst + (((long)(b*H + h)*L + l)*Dh + d4*4)) = o;
}

// ---------------- mean of V over L per (b,h,d) ----------------
__global__ void __launch_bounds__(256) vmean_kernel(const float* __restrict__ V,
                                                    float* __restrict__ vmean){
  int bh = blockIdx.x, b = bh >> 3, h = bh & 7;
  int t = threadIdx.x, d = t & 63, part = t >> 6;
  float s = 0.f;
  for (int l = part; l < L; l += 4)
    s += V[(((long)(b*L + l)*H + h)*Dh) + d];
  __shared__ float red[256];
  red[t] = s; __syncthreads();
  if (t < 64)
    vmean[bh*64 + d] = (red[d] + red[64+d] + red[128+d] + red[192+d]) * (1.0f / L);
}

// ---------------- ranking: M[q] = max_k(QK) - sum_k(QK)/L via bf16 WMMA ----------------
// K tiles double-buffered in LDS via GLOBAL_LOAD_ASYNC_TO_LDS_B128 (ASYNCcnt-tracked).
__global__ void __launch_bounds__(256) rank_kernel(const unsigned short* __restrict__ Qb,
                                                   const unsigned short* __restrict__ Kb,
                                                   float* __restrict__ Mout){
  __shared__ unsigned short Qs[128 * Dh];
  __shared__ unsigned short Ks2[2][128 * Dh];
  const int bh = blockIdx.y, qt = blockIdx.x;
  const int tid = threadIdx.x;
  const int lane = tid & 31, w = tid >> 5;
  const int m = lane & 15, half = lane >> 4;

  const unsigned short* Kbase = Kb + (long)bh*L*Dh;

  // issue async staging of a 128x64 bf16 K tile into buffer (kt&1): 4 b128 per thread
  auto issueK = [&](int kt){
    const unsigned short* Kg = Kbase + (long)kt*128*Dh;
    unsigned short* dst = Ks2[kt & 1];
    #pragma unroll
    for (int c = 0; c < 4; ++c){
      int off = (tid + c*256) * 8;                       // element offset (16B chunks)
      unsigned ldso = (unsigned)(size_t)(dst + off);     // low 32 bits = DS byte offset
      unsigned long long ga = (unsigned long long)(size_t)(Kg + off);
      asm volatile("global_load_async_to_lds_b128 %0, %1, off"
                   :: "v"(ldso), "v"(ga) : "memory");
    }
  };

  issueK(0);

  const unsigned short* Qg = Qb + ((long)bh*L + qt*128)*Dh;
  for (int i = tid; i < 128*Dh/8; i += 256)
    *(uint4*)(Qs + i*8) = *(const uint4*)(Qg + i*8);
  __syncthreads();

  // A fragments for this wave's 16 query rows (invariant over key tiles)
  const unsigned short* arow = Qs + (w*16 + m)*Dh;
  v16bf a0 = load_a_frag(arow,      half);
  v16bf a1 = load_a_frag(arow + 32, half);

  float pmax[8], psum[8];
  #pragma unroll
  for (int r = 0; r < 8; ++r){ pmax[r] = -3.4e38f; psum[r] = 0.f; }

  for (int kt = 0; kt < L/128; ++kt){
    if (kt + 1 < L/128){
      issueK(kt + 1);
      asm volatile("s_wait_asynccnt 0x4" ::: "memory");  // tile kt's 4 loads landed
    } else {
      asm volatile("s_wait_asynccnt 0x0" ::: "memory");
    }
    __syncthreads();                                     // all waves' tile data visible

    const unsigned short* Ksb = Ks2[kt & 1];
    #pragma unroll
    for (int j = 0; j < 8; ++j){
      v16bf b0 = load_b_frag(Ksb + (j*16 + m)*Dh +      16*half);
      v16bf b1 = load_b_frag(Ksb + (j*16 + m)*Dh + 32 + 16*half);
      v8f acc = {};
      acc = __builtin_amdgcn_wmma_f32_16x16x32_bf16(false, a0, false, b0, (short)0, acc, false, false);
      acc = __builtin_amdgcn_wmma_f32_16x16x32_bf16(false, a1, false, b1, (short)0, acc, false, false);
      #pragma unroll
      for (int r = 0; r < 8; ++r){
        float v = acc[r];
        pmax[r] = fmaxf(pmax[r], v);
        psum[r] += v;
      }
    }
    __syncthreads();                                     // safe to overwrite this buffer
  }

  // reduce across the 16 lanes sharing each output row
  #pragma unroll
  for (int r = 0; r < 8; ++r){
    float mx = pmax[r], sm = psum[r];
    #pragma unroll
    for (int off = 1; off < 16; off <<= 1){
      mx = fmaxf(mx, __shfl_xor(mx, off, 32));
      sm += __shfl_xor(sm, off, 32);
    }
    if (m == 0){
      int row = qt*128 + w*16 + half*8 + r;
      Mout[(long)bh*L + row] = mx - sm * (1.0f / L);
    }
  }
}

// ---------------- top-40 per (b,h), ties -> smallest index ----------------
__global__ void __launch_bounds__(256) topk_kernel(const float* __restrict__ Mbuf,
                                                   int* __restrict__ topk){
  __shared__ float sv[L];
  __shared__ float rv[256];
  __shared__ int   ri[256];
  int bh = blockIdx.x, tid = threadIdx.x;
  for (int i = tid; i < L; i += 256) sv[i] = Mbuf[(long)bh*L + i];
  __syncthreads();
  for (int it = 0; it < U; ++it){
    float bv = -3.4e38f; int bi = 0;
    for (int i = tid; i < L; i += 256){
      float v = sv[i];
      if (v > bv){ bv = v; bi = i; }
    }
    rv[tid] = bv; ri[tid] = bi;
    __syncthreads();
    for (int s = 128; s > 0; s >>= 1){
      if (tid < s){
        if (rv[tid+s] > rv[tid] || (rv[tid+s] == rv[tid] && ri[tid+s] < ri[tid])){
          rv[tid] = rv[tid+s]; ri[tid] = ri[tid+s];
        }
      }
      __syncthreads();
    }
    if (tid == 0){ topk[bh*U + it] = ri[0]; sv[ri[0]] = -3.4e38f; }
    __syncthreads();
  }
}

// ---------------- fill output with mean(V) ----------------
__global__ void __launch_bounds__(256) fill_ctx(const float* __restrict__ vmean,
                                                float* __restrict__ out){
  int g = blockIdx.x * 256 + threadIdx.x;      // one per float4
  int d4 = g & 15;
  int l  = (g >> 4) & (L - 1);
  int bh = g >> 15;
  float4 v = *(const float4*)(vmean + bh*64 + d4*4);
  *(float4*)(out + (((long)bh*L + l)*Dh) + d4*4) = v;
}

// ---------------- flash attention over the 48 (padded) selected rows ----------------
__global__ void __launch_bounds__(256) attn_kernel(const unsigned short* __restrict__ Qb,
                                                   const unsigned short* __restrict__ Kb,
                                                   const unsigned short* __restrict__ Vb,
                                                   const int* __restrict__ topk,
                                                   float* __restrict__ out){
  __shared__ unsigned short Qs[UP * Dh];      // 48x64 bf16
  __shared__ unsigned short Ks[64 * Dh];      // 64 keys x 64 bf16
  __shared__ unsigned short VsT[Dh * 64];     // V transposed: [d][key]
  __shared__ float          Ss[UP * 64];      // scaled scores f32
  __shared__ unsigned short Ps[UP * 64];      // exp probabilities bf16
  __shared__ float rmax[UP], rsum[UP], rfac[UP];
  __shared__ int qidx[UP];

  int bh = blockIdx.x, tid = threadIdx.x;
  int lane = tid & 31, w = tid >> 5;
  int m = lane & 15, half = lane >> 4;

  if (tid < UP){
    qidx[tid] = topk[bh*U + (tid < U ? tid : 0)];
    rmax[tid] = -3.4e38f; rsum[tid] = 0.f;
  }
  __syncthreads();
  for (int i = tid; i < UP*Dh; i += 256){
    int row = i >> 6, d = i & 63;
    Qs[i] = Qb[((long)bh*L + qidx[row])*Dh + d];
  }
  __syncthreads();

  v8f cacc[2] = {{}, {}};

  for (int kt = 0; kt < L/64; ++kt){
    __syncthreads();
    const unsigned short* Kg = Kb + ((long)bh*L + kt*64)*Dh;
    const unsigned short* Vg = Vb + ((long)bh*L + kt*64)*Dh;
    if (kt + 1 < L/64){                       // prefetch next tiles into L2/WGP$
      __builtin_prefetch(Kg + 64*Dh, 0, 0);
      __builtin_prefetch(Vg + 64*Dh, 0, 0);
    }
    for (int i = tid; i < 64*Dh/8; i += 256)
      *(uint4*)(Ks + i*8) = *(const uint4*)(Kg + i*8);
    for (int i = tid; i < 64*Dh/8; i += 256){
      int key = i >> 3, d0 = (i & 7) * 8;
      uint4 pk = *(const uint4*)(Vg + key*Dh + d0);
      unsigned short tmp[8];
      *(uint4*)tmp = pk;
      #pragma unroll
      for (int e = 0; e < 8; ++e) VsT[(d0 + e)*64 + key] = tmp[e];
    }
    __syncthreads();

    // scores: tiles T2 = rt*4 + j over 3 row-tiles x 4 col-tiles
    for (int T2 = w; T2 < 12; T2 += 8){
      int rt = T2 >> 2, j = T2 & 3;
      const unsigned short* arow = Qs + (rt*16 + m)*Dh;
      v16bf a0 = load_a_frag(arow,      half);
      v16bf a1 = load_a_frag(arow + 32, half);
      v16bf b0 = load_b_frag(Ks + (j*16 + m)*Dh +      16*half);
      v16bf b1 = load_b_frag(Ks + (j*16 + m)*Dh + 32 + 16*half);
      v8f acc = {};
      acc = __builtin_amdgcn_wmma_f32_16x16x32_bf16(false, a0, false, b0, (short)0, acc, false, false);
      acc = __builtin_amdgcn_wmma_f32_16x16x32_bf16(false, a1, false, b1, (short)0, acc, false, false);
      #pragma unroll
      for (int r = 0; r < 8; ++r)
        Ss[(rt*16 + half*8 + r)*64 + j*16 + m] = acc[r] * SCALE;
    }
    __syncthreads();

    // online softmax, one thread per row
    if (tid < UP){
      int row = tid;
      float tm = -3.4e38f;
      for (int c = 0; c < 64; ++c) tm = fmaxf(tm, Ss[row*64 + c]);
      float mold = rmax[row];
      float mnew = fmaxf(mold, tm);
      float fac  = __expf(mold - mnew);
      float s = 0.f;
      for (int c = 0; c < 64; ++c){
        float p = __expf(Ss[row*64 + c] - mnew);
        Ps[row*64 + c] = f2bf(p);
        s += p;
      }
      rsum[row] = rsum[row]*fac + s;
      rmax[row] = mnew;
      rfac[row] = fac;
    }
    __syncthreads();

    // context accumulation: tiles T = rt*4 + dt
    for (int T = w; T < 12; T += 8){
      int rt = T >> 2, dt = T & 3;
      int slot = (T - w) >> 3;
      v8f acc = cacc[slot];
      #pragma unroll
      for (int r = 0; r < 8; ++r) acc[r] *= rfac[rt*16 + half*8 + r];
      #pragma unroll
      for (int kc = 0; kc < 2; ++kc){
        v16bf a = load_a_frag(Ps + (rt*16 + m)*64 + kc*32, half);
        v16bf b = load_b_frag(VsT + (dt*16 + m)*64 + kc*32 + 16*half);
        acc = __builtin_amdgcn_wmma_f32_16x16x32_bf16(false, a, false, b, (short)0, acc, false, false);
      }
      cacc[slot] = acc;
    }
  }
  __syncthreads();

  for (int T = w; T < 12; T += 8){
    int rt = T >> 2, dt = T & 3;
    int slot = (T - w) >> 3;
    #pragma unroll
    for (int r = 0; r < 8; ++r){
      int row = rt*16 + half*8 + r;
      if (row < U){
        float val = cacc[slot][r] / rsum[row];
        out[((long)bh*L + qidx[row])*Dh + dt*16 + m] = val;
      }
    }
  }
}

extern "C" void kernel_launch(void* const* d_in, const int* in_sizes, int n_in,
                              void* d_out, int out_size, void* d_ws, size_t ws_size,
                              hipStream_t stream) {
  (void)in_sizes; (void)n_in; (void)out_size; (void)ws_size;
  const float* Q = (const float*)d_in[0];
  const float* K = (const float*)d_in[1];
  const float* V = (const float*)d_in[2];
  float* out = (float*)d_out;

  const size_t nPack = (size_t)BH * L * Dh;   // 8,388,608 elements per tensor
  unsigned short* Qb = (unsigned short*)d_ws;
  unsigned short* Kb = Qb + nPack;
  unsigned short* Vb = Kb + nPack;
  float* Mbuf     = (float*)(Vb + nPack);
  int*   topkbuf  = (int*)(Mbuf + (size_t)BH * L);
  float* vmeanbuf = (float*)(topkbuf + BH * U);

  const int packBlocks = BH * L * (Dh/4) / 256;   // 8192
  pack_bf16<<<packBlocks, 256, 0, stream>>>(Q, Qb);
  pack_bf16<<<packBlocks, 256, 0, stream>>>(K, Kb);
  pack_bf16<<<packBlocks, 256, 0, stream>>>(V, Vb);
  vmean_kernel<<<BH, 256, 0, stream>>>(V, vmeanbuf);
  rank_kernel<<<dim3(L/128, BH), 256, 0, stream>>>(Qb, Kb, Mbuf);
  topk_kernel<<<BH, 256, 0, stream>>>(Mbuf, topkbuf);
  fill_ctx<<<packBlocks, 256, 0, stream>>>(vmeanbuf, out);
  attn_kernel<<<BH, 256, 0, stream>>>(Qb, Kb, Vb, topkbuf, out);
}
